// TitansMACMemory_1477468750476
// MI455X (gfx1250) — compile-verified, matrix operand order checked by reading, not run on hardware
//
#include <hip/hip_runtime.h>
#include <math.h>

// ---------------------------------------------------------------- types ----
typedef __bf16 bf16;
typedef bf16  v16bf __attribute__((ext_vector_type(16)));
typedef bf16  v8bf  __attribute__((ext_vector_type(8)));
typedef float v8f   __attribute__((ext_vector_type(8)));

// ------------------------------------------------------------- problem ----
constexpr int BB = 32, TT = 1024, HH = 1024, DD = 64;
constexpr int BT = BB * TT;                       // 32768 rows

constexpr float ETA = 0.1f, BETA = 0.9f, ALPHA = 0.99f;
constexpr float INV_SQRT2   = 0.70710678118654752440f;
constexpr float INV_SQRT2PI = 0.3989422804014327f;

__device__ __forceinline__ float gelu_f(float x) {
    return 0.5f * x * (1.0f + erff(x * INV_SQRT2));
}
__device__ __forceinline__ float gelu_grad_f(float x) {
    float cdf = 0.5f * (1.0f + erff(x * INV_SQRT2));
    float pdf = expf(-0.5f * x * x) * INV_SQRT2PI;
    return cdf + x * pdf;
}

__device__ __forceinline__ v16bf make16(v8bf lo, v8bf hi) {
    v16bf r;
#pragma unroll
    for (int i = 0; i < 8; ++i) { r[i] = lo[i]; r[i + 8] = hi[i]; }
    return r;
}

// ---- CDNA5 async global->LDS copy (ASYNCcnt path), 16B per lane ----------
__device__ __forceinline__ void async_b128(uint32_t lds_addr, uint64_t gaddr) {
    asm volatile("global_load_async_to_lds_b128 %0, %1, off"
                 :: "v"(lds_addr), "v"(gaddr) : "memory");
}
__device__ __forceinline__ void async_b128_off16(uint32_t lds_addr, uint64_t gaddr) {
    asm volatile("global_load_async_to_lds_b128 %0, %1, off offset:16"
                 :: "v"(lds_addr), "v"(gaddr) : "memory");
}
__device__ __forceinline__ void wait_async0() {
#if __has_builtin(__builtin_amdgcn_s_wait_asynccnt)
    __builtin_amdgcn_s_wait_asynccnt(0);
#else
    asm volatile("s_wait_asynccnt 0x0" ::: "memory");
#endif
}

// ------------------------------------------------------- cast / copy -------
__global__ __launch_bounds__(256) void cast_f32_to_bf16(const float* __restrict__ in,
                                                        bf16* __restrict__ out, size_t n) {
    size_t i = (size_t)blockIdx.x * blockDim.x + threadIdx.x;
    if (i < n) out[i] = (bf16)in[i];
}
__global__ __launch_bounds__(256) void copy_f32(const float* __restrict__ in,
                                                float* __restrict__ out, size_t n) {
    size_t i = (size_t)blockIdx.x * blockDim.x + threadIdx.x;
    if (i < n) out[i] = in[i];
}

// ------------------------------------------- WMMA GEMM:  C = A * B^T ------
// A: MxK bf16 row-major (lda = K), B: NxK bf16 row-major (ldb = K),
// C: MxN f32 (ldc = N).  Reduction dim contiguous for both operands.
// Block: 256 thr = 8 wave32.  Block tile 128x64, wave tile 32x32 (2x2 WMMA).
// Staging: double-buffered global_load_async_to_lds_b128 (ASYNCcnt).
// EPI: 0 = f32 out, 1 = f32 + bf16 out, 2 = gelu -> bf16 only.
constexpr int TM = 128, TN = 64, TK = 32, LDSW = 40;   // LDSW padded vs 32

template <int EPI>
__global__ __launch_bounds__(256)
void gemm_abT_wmma(const bf16* __restrict__ A, const bf16* __restrict__ Bm,
                   float* __restrict__ C, bf16* __restrict__ Cb,
                   int N, int K) {
    __shared__ __align__(16) bf16 lA[2][TM * LDSW];
    __shared__ __align__(16) bf16 lB[2][TN * LDSW];

    const int tid  = threadIdx.x;
    const int wave = tid >> 5, lane = tid & 31;
    const int waveM = (wave >> 1) * 32;       // 0,32,64,96
    const int waveN = (wave & 1) * 32;        // 0,32
    const size_t m0 = (size_t)blockIdx.x * TM;
    const size_t n0 = (size_t)blockIdx.y * TN;

    v8f acc[2][2];
#pragma unroll
    for (int s = 0; s < 2; ++s)
#pragma unroll
        for (int u = 0; u < 2; ++u)
#pragma unroll
            for (int r = 0; r < 8; ++r) acc[s][u][r] = 0.0f;

    const int ar = tid >> 1, ach = (tid & 1) * 16;   // A tile: 128 rows x 2 x 16-elem chunks
    const int br = tid >> 2, bch = (tid & 3) * 8;    // B tile:  64 rows x 4 x 8-elem chunks
    const int fr = lane & 15, fk = (lane >> 4) * 8;  // fragment row / k-half

    // issue one K-slab of async copies into LDS buffer `bufi`
    auto issue_tile = [&](int kt, int bufi) {
        uint64_t gaA = (uint64_t)(uintptr_t)(A  + (m0 + ar) * (size_t)K + kt + ach);
        uint32_t laA = (uint32_t)(uintptr_t)&lA[bufi][ar * LDSW + ach];
        async_b128(laA, gaA);           // elements [ach, ach+8)
        async_b128_off16(laA, gaA);     // elements [ach+8, ach+16)
        uint64_t gaB = (uint64_t)(uintptr_t)(Bm + (n0 + br) * (size_t)K + kt + bch);
        uint32_t laB = (uint32_t)(uintptr_t)&lB[bufi][br * LDSW + bch];
        async_b128(laB, gaB);
    };

    int buf = 0;
    issue_tile(0, 0);
    for (int kt = 0; kt < K; kt += TK) {
        wait_async0();        // my async copies (into `buf`) have landed
        __syncthreads();      // everyone's have landed; prior readers done
        if (kt + TK < K) issue_tile(kt + TK, buf ^ 1);

        v16bf aF[2], bF[2];
#pragma unroll
        for (int s = 0; s < 2; ++s) {
            int row = waveM + s * 16 + fr;
            v8bf lo = *(const v8bf*)&lA[buf][row * LDSW + fk];
            v8bf hi = *(const v8bf*)&lA[buf][row * LDSW + fk + 16];
            aF[s] = make16(lo, hi);
        }
#pragma unroll
        for (int u = 0; u < 2; ++u) {
            int row = waveN + u * 16 + fr;
            v8bf lo = *(const v8bf*)&lB[buf][row * LDSW + fk];
            v8bf hi = *(const v8bf*)&lB[buf][row * LDSW + fk + 16];
            bF[u] = make16(lo, hi);
        }
#pragma unroll
        for (int s = 0; s < 2; ++s)
#pragma unroll
            for (int u = 0; u < 2; ++u)
                acc[s][u] = __builtin_amdgcn_wmma_f32_16x16x32_bf16(
                    false, aF[s], false, bF[u], (short)0, acc[s][u], false, false);
        buf ^= 1;
    }

    // epilogue: C/D layout = VGPR r -> (M = r | r+8 by lane-half, N = lane&15)
    const int cn    = lane & 15;
    const int rbase = (lane < 16) ? 0 : 8;
#pragma unroll
    for (int s = 0; s < 2; ++s)
#pragma unroll
        for (int u = 0; u < 2; ++u)
#pragma unroll
            for (int r = 0; r < 8; ++r) {
                size_t row = m0 + waveM + s * 16 + rbase + r;
                size_t col = n0 + waveN + u * 16 + cn;
                float  v   = acc[s][u][r];
                size_t idx = row * (size_t)N + col;
                if constexpr (EPI == 2) {
                    Cb[idx] = (bf16)gelu_f(v);
                } else {
                    C[idx] = v;
                    if constexpr (EPI == 1) Cb[idx] = (bf16)v;
                }
            }
}

// --------------------------------------------------- assoc loss (per t) ---
__global__ __launch_bounds__(256)
void assoc_loss_kernel(const float* __restrict__ reads, const float* __restrict__ Vf,
                       float* __restrict__ assoc) {
    const int t = blockIdx.x, tid = threadIdx.x;
    __shared__ float red[256];
    float s = 0.0f;
    for (int b = 0; b < BB; ++b) {
        const float* r = reads + ((size_t)b * TT + t) * HH;
        const float* v = Vf    + ((size_t)b * TT + t) * HH;
        for (int hh = tid; hh < HH; hh += 256) {
            float d = r[hh] - v[hh];
            s = fmaf(d, d, s);
        }
    }
    red[tid] = s; __syncthreads();
    for (int st = 128; st > 0; st >>= 1) {
        if (tid < st) red[tid] += red[tid + st];
        __syncthreads();
    }
    if (tid == 0) assoc[t] = red[0] / (float)(BB * HH);
}

// --------------------------------------------- sequential Titans scan -----
// Single workgroup (32 wave32s).  Working set ~2 MB -> L2-resident.
__global__ __launch_bounds__(1024)
void titans_scan(const float* __restrict__ Kf, const float* __restrict__ Vf,
                 float* __restrict__ W1c, float* __restrict__ W2c,
                 float* __restrict__ m1c, float* __restrict__ m2c,
                 float* __restrict__ hbuf, float* __restrict__ abuf,
                 float* __restrict__ dpredbuf, float* __restrict__ dhbuf,
                 float* __restrict__ g1buf, float* __restrict__ g2buf,
                 float* __restrict__ surp) {
    const int tid = threadIdx.x;
    __shared__ float red1[1024];
    __shared__ float red2[1024];
    __shared__ float sc[2];
    const float two_over_denom = 2.0f / (float)(BB * HH);

    for (int t = 0; t < TT; ++t) {
        // phase 1: h = k_t @ W1^T ; a = gelu(h)           (32x64, dot 1024)
        for (int o = tid; o < BB * DD; o += 1024) {
            int i = o >> 6, d = o & 63;
            const float* krow = Kf  + ((size_t)i * TT + t) * HH;
            const float* wrow = W1c + (size_t)d * HH;
            float s = 0.0f;
            for (int hh = 0; hh < HH; ++hh) s = fmaf(krow[hh], wrow[hh], s);
            hbuf[o] = s; abuf[o] = gelu_f(s);
        }
        __syncthreads();
        // phase 2: dpred = (2/denom)*(a @ W2^T - v_t)     (32x1024, dot 64)
        for (int o = tid; o < BB * HH; o += 1024) {
            int i = o >> 10, hh = o & 1023;
            const float* arow = abuf + i * DD;
            const float* wrow = W2c  + (size_t)hh * DD;
            float s = 0.0f;
            for (int d = 0; d < DD; ++d) s = fmaf(arow[d], wrow[d], s);
            float v = Vf[((size_t)i * TT + t) * HH + hh];
            dpredbuf[o] = two_over_denom * (s - v);
        }
        __syncthreads();
        // phase 3: g2 = dpred^T @ a                       (1024x64, dot 32)
        for (int o = tid; o < HH * DD; o += 1024) {
            int hh = o >> 6, d = o & 63;
            float s = 0.0f;
            for (int i = 0; i < BB; ++i)
                s = fmaf(dpredbuf[i * HH + hh], abuf[i * DD + d], s);
            g2buf[o] = s;
        }
        // phase 4: dh = (dpred @ W2) * gelu'(h)           (32x64, dot 1024)
        for (int o = tid; o < BB * DD; o += 1024) {
            int i = o >> 6, d = o & 63;
            const float* drow = dpredbuf + i * HH;
            float s = 0.0f;
            for (int hh = 0; hh < HH; ++hh)
                s = fmaf(drow[hh], W2c[(size_t)hh * DD + d], s);
            dhbuf[o] = s * gelu_grad_f(hbuf[o]);
        }
        __syncthreads();
        // phase 5: g1 = dh^T @ k_t                        (64x1024, dot 32)
        for (int o = tid; o < DD * HH; o += 1024) {
            int d = o >> 10, hh = o & 1023;
            float s = 0.0f;
            for (int i = 0; i < BB; ++i)
                s = fmaf(dhbuf[i * DD + d], Kf[((size_t)i * TT + t) * HH + hh], s);
            g1buf[o] = s;
        }
        __syncthreads();
        // phase 6: Frobenius norms, surprise, clip scales
        float s1 = 0.0f, s2 = 0.0f;
        for (int o = tid; o < DD * HH; o += 1024) {
            float a = g1buf[o]; s1 = fmaf(a, a, s1);
            float b = g2buf[o]; s2 = fmaf(b, b, s2);
        }
        red1[tid] = s1; red2[tid] = s2; __syncthreads();
        for (int st = 512; st > 0; st >>= 1) {
            if (tid < st) { red1[tid] += red1[tid + st]; red2[tid] += red2[tid + st]; }
            __syncthreads();
        }
        if (tid == 0) {
            float n1 = sqrtf(red1[0]), n2 = sqrtf(red2[0]);
            surp[t] = 0.5f * (n1 + n2);
            sc[0] = (n1 > 1.0f) ? 1.0f / n1 : 1.0f;
            sc[1] = (n2 > 1.0f) ? 1.0f / n2 : 1.0f;
        }
        __syncthreads();
        const float sc1 = sc[0], sc2 = sc[1];
        // phase 7: momentum + weight decay updates
        for (int o = tid; o < DD * HH; o += 1024) {
            float m1v = BETA * m1c[o] - ETA * (g1buf[o] * sc1);
            m1c[o] = m1v; W1c[o] = ALPHA * W1c[o] + m1v;
            float m2v = BETA * m2c[o] - ETA * (g2buf[o] * sc2);
            m2c[o] = m2v; W2c[o] = ALPHA * W2c[o] + m2v;
        }
        __syncthreads();     // before next step overwrites scratch
    }
}

// ------------------------------------------------------------- launch -----
extern "C" void kernel_launch(void* const* d_in, const int* in_sizes, int n_in,
                              void* d_out, int out_size, void* d_ws, size_t ws_size,
                              hipStream_t stream) {
    const float* x  = (const float*)d_in[0];
    const float* WK = (const float*)d_in[1];
    const float* WV = (const float*)d_in[2];
    const float* W1 = (const float*)d_in[3];
    const float* W2 = (const float*)d_in[4];
    const float* m1 = (const float*)d_in[5];
    const float* m2 = (const float*)d_in[6];

    float* out       = (float*)d_out;
    float* reads_out = out;
    float* surp_out  = out + (size_t)BT * HH;
    float* assoc_out = surp_out + TT;
    float* w1f_out   = assoc_out + TT;
    float* w2f_out   = w1f_out + (size_t)DD * HH;

    char*  ws = (char*)d_ws;
    size_t off = 0;
    auto alloc = [&](size_t bytes) -> char* {
        char* p = ws + off;
        off = (off + bytes + 255) & ~(size_t)255;
        return p;
    };
    bf16*  xb   = (bf16*) alloc((size_t)BT * HH * 2);
    bf16*  wkb  = (bf16*) alloc((size_t)HH * HH * 2);
    bf16*  wvb  = (bf16*) alloc((size_t)HH * HH * 2);
    bf16*  w1b  = (bf16*) alloc((size_t)DD * HH * 2);
    bf16*  w2b  = (bf16*) alloc((size_t)HH * DD * 2);
    float* Kf   = (float*)alloc((size_t)BT * HH * 4);
    float* Vf   = (float*)alloc((size_t)BT * HH * 4);
    bf16*  Kb   = (bf16*) alloc((size_t)BT * HH * 2);
    bf16*  hidb = (bf16*) alloc((size_t)BT * DD * 2);
    float* W1c  = (float*)alloc((size_t)DD * HH * 4);
    float* W2c  = (float*)alloc((size_t)HH * DD * 4);
    float* m1c  = (float*)alloc((size_t)DD * HH * 4);
    float* m2c  = (float*)alloc((size_t)HH * DD * 4);
    float* hbuf = (float*)alloc((size_t)BB * DD * 4);
    float* abuf = (float*)alloc((size_t)BB * DD * 4);
    float* dpb  = (float*)alloc((size_t)BB * HH * 4);
    float* dhb  = (float*)alloc((size_t)BB * DD * 4);
    float* g1b  = (float*)alloc((size_t)DD * HH * 4);
    float* g2b  = (float*)alloc((size_t)HH * DD * 4);

    auto blocks = [](size_t n) { return (unsigned)((n + 255) / 256); };

    // precision staging
    cast_f32_to_bf16<<<blocks((size_t)BT * HH), 256, 0, stream>>>(x,  xb,  (size_t)BT * HH);
    cast_f32_to_bf16<<<blocks((size_t)HH * HH), 256, 0, stream>>>(WK, wkb, (size_t)HH * HH);
    cast_f32_to_bf16<<<blocks((size_t)HH * HH), 256, 0, stream>>>(WV, wvb, (size_t)HH * HH);
    cast_f32_to_bf16<<<blocks((size_t)DD * HH), 256, 0, stream>>>(W1, w1b, (size_t)DD * HH);
    cast_f32_to_bf16<<<blocks((size_t)HH * DD), 256, 0, stream>>>(W2, w2b, (size_t)HH * DD);
    copy_f32<<<blocks((size_t)DD * HH), 256, 0, stream>>>(W1, W1c, (size_t)DD * HH);
    copy_f32<<<blocks((size_t)HH * DD), 256, 0, stream>>>(W2, W2c, (size_t)HH * DD);
    copy_f32<<<blocks((size_t)DD * HH), 256, 0, stream>>>(m1, m1c, (size_t)DD * HH);
    copy_f32<<<blocks((size_t)HH * DD), 256, 0, stream>>>(m2, m2c, (size_t)HH * DD);

    // K = x @ W_K^T (f32 + bf16), V = x @ W_V^T (f32)
    gemm_abT_wmma<1><<<dim3(BT / TM, HH / TN), 256, 0, stream>>>(xb, wkb, Kf, Kb,      HH, HH);
    gemm_abT_wmma<0><<<dim3(BT / TM, HH / TN), 256, 0, stream>>>(xb, wvb, Vf, nullptr, HH, HH);
    // hidden = gelu(K @ W1^T)  (bf16 only)
    gemm_abT_wmma<2><<<dim3(BT / TM, DD / TN), 256, 0, stream>>>(Kb, w1b, nullptr, hidb, DD, HH);
    // reads = hidden @ W2^T  -> d_out
    gemm_abT_wmma<0><<<dim3(BT / TM, HH / TN), 256, 0, stream>>>(hidb, w2b, reads_out, nullptr, HH, DD);

    // per-token assoc losses
    assoc_loss_kernel<<<TT, 256, 0, stream>>>(reads_out, Vf, assoc_out);

    // sequential Titans update scan (single WGP-resident workgroup)
    titans_scan<<<1, 1024, 0, stream>>>(Kf, Vf, W1c, W2c, m1c, m2c,
                                        hbuf, abuf, dpb, dhb, g1b, g2b, surp_out);

    // final weights -> d_out
    copy_f32<<<blocks((size_t)DD * HH), 256, 0, stream>>>(W1c, w1f_out, (size_t)DD * HH);
    copy_f32<<<blocks((size_t)HH * DD), 256, 0, stream>>>(W2c, w2f_out, (size_t)HH * DD);
}
